// Cca3_2748779070023
// MI455X (gfx1250) — compile-verified, matrix operand order checked by reading, not run on hardware
//
#include <hip/hip_runtime.h>
#include <hip/hip_bf16.h>

#define CDIM 128
#define NPIX 16384   // 128*128 spatial
#define NB   16      // total batches
#define HB   8       // half (s / t split)
#define LDST 136     // padded LDS row stride in bf16 elems (272B, 16B aligned)

typedef __attribute__((ext_vector_type(16))) __bf16 v16bf;
typedef __attribute__((ext_vector_type(8)))  float  v8f;
typedef __attribute__((ext_vector_type(4))) unsigned v4u;
typedef __attribute__((ext_vector_type(8))) int      v8i;
typedef __attribute__((address_space(1))) __bf16 gbf16;   // force global stores
typedef __attribute__((address_space(3))) char   lds_char;

// Load one 16x32 bf16 WMMA operand fragment from a row-major matrix.
// p points at: row_base + k0 + 8*(lane>>4).  elems 0..7 at p, elems 8..15 at p+16.
__device__ __forceinline__ v16bf frag_ld(const __bf16* p) {
  v16bf f;
  ((uint4*)&f)[0] = *(const uint4*)(p);
  ((uint4*)&f)[1] = *(const uint4*)(p + 16);
  return f;
}

__device__ __forceinline__ v8f wmma_bf16(v16bf a, v16bf b, v8f c) {
  return __builtin_amdgcn_wmma_f32_16x16x32_bf16(false, a, false, b, (short)0, c,
                                                 false, false);
}

// 32-bit LDS byte offset of a generic pointer to __shared__ memory
__device__ __forceinline__ unsigned lds_addr(void* p) {
  return (unsigned)(__SIZE_TYPE__)(lds_char*)p;
}

// CDNA5 async copy: 16B global -> LDS, tracked by ASYNCcnt (no VGPR round trip)
__device__ __forceinline__ void async_cp_b128(unsigned lds_dst, const void* gsrc) {
  asm volatile("global_load_async_to_lds_b128 %0, %1, off"
               :: "v"(lds_dst), "v"((unsigned long long)(__SIZE_TYPE__)gsrc)
               : "memory");
}
__device__ __forceinline__ void async_wait0() {
  asm volatile("s_wait_asynccnt 0x0" ::: "memory");
}

// CDNA5 Tensor Data Mover: 2D tile (rows x 128 bf16 elems, row stride `stride`
// elems) global -> LDS, +16B LDS padding per 256B row (matches LDST=136).
// Uniform args only (descriptor lives in SGPRs).  Tracked by TENSORcnt.
__device__ __forceinline__ void tdm_load_tile_2d(unsigned lds_byte,
                                                 const void* gsrc,
                                                 unsigned rows, unsigned stride) {
  unsigned long long ga = (unsigned long long)(__SIZE_TYPE__)gsrc;
  v4u g0;
  g0[0] = 1u;                                   // count=1 (valid user descriptor)
  g0[1] = lds_byte;                             // lds_addr
  g0[2] = (unsigned)ga;                         // global_addr[31:0]
  g0[3] = (unsigned)(ga >> 32) | (2u << 30);    // global_addr[56:32] | type=2
  v8i g1;
  g1[0] = (int)((1u << 16)      // data_size = 1 (2 bytes)
              | (1u << 20)      // pad_enable
              | (5u << 22)      // pad_interval: 64 DWORDs (=256B row)
              | (3u << 25));    // pad_amount: 4 DWORDs (=16B)
  g1[1] = (int)((NPIX & 0xFFFFu) << 16);        // tensor_dim0[15:0] (=16384)
  g1[2] = (int)(((NPIX >> 16) & 0xFFFFu) | ((rows & 0xFFFFu) << 16)); // td0 hi, td1 lo
  g1[3] = (int)(128u << 16);                    // td1 hi=0 | tile_dim0=128
  g1[4] = (int)(rows & 0xFFFFu);                // tile_dim1=rows | tile_dim2=0
  g1[5] = (int)stride;                          // tensor_dim0_stride[31:0]
  g1[6] = 0;                                    // stride hi | dim1_stride lo
  g1[7] = 0;                                    // dim1_stride hi
  asm volatile("tensor_load_to_lds %0, %1" :: "s"(g0), "s"(g1) : "memory");
}

// ---------------------------------------------------------------- zero gram
__global__ void k_zero(float* __restrict__ s, int n) {
  int i = blockIdx.x * blockDim.x + threadIdx.x;
  if (i < n) s[i] = 0.f;
}

// ---------------------------------------------------------------- projections
// grid: (NPIX/128, NB), block 256 (8 waves).  Q/K/V[b][o][n] bf16.
__global__ void __launch_bounds__(256)
k_proj(const float* __restrict__ x,
       const float* __restrict__ qw, const float* __restrict__ qbias,
       const float* __restrict__ kw, const float* __restrict__ kbias,
       const float* __restrict__ vw, const float* __restrict__ vbias,
       __bf16* __restrict__ Q, __bf16* __restrict__ K, __bf16* __restrict__ V) {
  extern __shared__ __align__(16) char smem[];
  __bf16* sW  = (__bf16*)smem;               // [3][128][LDST]  weights bf16
  __bf16* sXT = sW + 3 * CDIM * LDST;        // [128 n][LDST]   x tile transposed
  float*  sBias = (float*)(sXT + CDIM * LDST); // [3][128]

  const int tid   = threadIdx.x;
  const int batch = blockIdx.y;
  const int n0    = blockIdx.x * 128;

  const float* Ws[3] = {qw, kw, vw};
  const float* Bs[3] = {qbias, kbias, vbias};
  __bf16*      Os[3] = {Q, K, V};

  for (int p = 0; p < 3; ++p)
    for (int i = tid; i < CDIM * CDIM; i += 256) {
      int o = i >> 7, c = i & 127;
      sW[p * CDIM * LDST + o * LDST + c] = (__bf16)Ws[p][i];
    }
  if (tid < 3 * CDIM) sBias[tid] = Bs[tid >> 7][tid & 127];
  const float* xb = x + (size_t)batch * CDIM * NPIX;
  for (int i = tid; i < CDIM * 128; i += 256) {
    int c = i >> 7, j = i & 127;
    sXT[j * LDST + c] = (__bf16)xb[(size_t)c * NPIX + n0 + j];   // transpose
  }
  __syncthreads();

  const int wave = tid >> 5, lane = tid & 31, half = lane >> 4, lid = lane & 15;
  for (int p = 0; p < 3; ++p) {
    const __bf16* ab = sW + p * CDIM * LDST + (wave * 16 + lid) * LDST + 8 * half;
    v16bf a0 = frag_ld(ab), a1 = frag_ld(ab + 32), a2 = frag_ld(ab + 64),
          a3 = frag_ld(ab + 96);
    gbf16* outp = (gbf16*)(Os[p] + (size_t)batch * CDIM * NPIX);
    for (int nt = 0; nt < 8; ++nt) {
      const __bf16* bb = sXT + (nt * 16 + lid) * LDST + 8 * half;
      v8f acc = {};
      acc = wmma_bf16(a0, frag_ld(bb), acc);
      acc = wmma_bf16(a1, frag_ld(bb + 32), acc);
      acc = wmma_bf16(a2, frag_ld(bb + 64), acc);
      acc = wmma_bf16(a3, frag_ld(bb + 96), acc);
      int col = n0 + nt * 16 + lid;
#pragma unroll
      for (int i = 0; i < 8; ++i) {
        int o = wave * 16 + half * 8 + i;              // C layout: M = i + 8*half
        outp[(size_t)o * NPIX + col] = (__bf16)(acc[i] + sBias[p * CDIM + o]);
      }
    }
  }
}

// ---------------------------------------------------------------- gram (split-K)
// grid: (16 n-chunks, 8 batches, 2 matrices). S[z*8+b][c][d] += sum_n Q[c,n]K[d,n]
// Tile staging done by the Tensor Data Mover (wave 0 issues, TENSORcnt waits).
__global__ void __launch_bounds__(256)
k_gram(const __bf16* __restrict__ Q, const __bf16* __restrict__ K,
       float* __restrict__ S) {
  extern __shared__ __align__(16) char smem[];
  __bf16* sQ = (__bf16*)smem;            // [128][LDST]
  __bf16* sK = sQ + CDIM * LDST;

  const int tid = threadIdx.x;
  const int b = blockIdx.y, z = blockIdx.z;
  const __bf16* Qb = Q + (size_t)(z == 0 ? b + HB : b) * CDIM * NPIX;  // q_t / q_s
  const __bf16* Kb = K + (size_t)(z == 0 ? b : b + HB) * CDIM * NPIX;  // k_s / k_t
  float* Sb = S + (size_t)(z * HB + b) * CDIM * CDIM;

  const unsigned sQ0 = lds_addr(sQ), sK0 = lds_addr(sK);
  const int wave = tid >> 5, lane = tid & 31, half = lane >> 4, lid = lane & 15;
  v8f acc[8];
#pragma unroll
  for (int dt = 0; dt < 8; ++dt) acc[dt] = (v8f){};

  for (int ns = 0; ns < 8; ++ns) {
    int n0 = blockIdx.x * 1024 + ns * 128;
    __syncthreads();                        // everyone done with previous tiles
    if (wave == 0) {                        // TDM: DMA both 128x128-elem tiles
      tdm_load_tile_2d(sQ0, Qb + n0, CDIM, NPIX);
      tdm_load_tile_2d(sK0, Kb + n0, CDIM, NPIX);
      __builtin_amdgcn_s_wait_tensorcnt(0);
    }
    __syncthreads();                        // tiles visible to all waves
#pragma unroll
    for (int ks = 0; ks < 4; ++ks) {
      v16bf a = frag_ld(sQ + (wave * 16 + lid) * LDST + ks * 32 + 8 * half);
#pragma unroll
      for (int dt = 0; dt < 8; ++dt) {
        v16bf bf = frag_ld(sK + (dt * 16 + lid) * LDST + ks * 32 + 8 * half);
        acc[dt] = wmma_bf16(a, bf, acc[dt]);
      }
    }
  }
#pragma unroll
  for (int dt = 0; dt < 8; ++dt) {
    int d = dt * 16 + lid;
#pragma unroll
    for (int i = 0; i < 8; ++i) {
      int c = wave * 16 + half * 8 + i;
      unsafeAtomicAdd(&Sb[c * CDIM + d], acc[dt][i]);   // global_atomic_add_f32
    }
  }
}

// ---------------------------------------------------------------- attention core
// grid: 8 blocks (one per batch pair), block 256.
__global__ void __launch_bounds__(256)
k_attn(const float* __restrict__ S, __bf16* __restrict__ att) {
  extern __shared__ __align__(16) char smem[];
  __bf16* sA = (__bf16*)smem;                  // softmax(S_st) [128][LDST]
  __bf16* sB = sA + CDIM * LDST;               // softmax(S_ts)
  float*  sT = (float*)(sB + CDIM * LDST);     // [128][129]
  const int TS = 129;

  const int tid = threadIdx.x;
  const int b = blockIdx.x;
  // row softmax of both gram matrices -> bf16 in LDS
  {
    const float* src = (tid < 128) ? (S + (size_t)b * CDIM * CDIM)
                                   : (S + (size_t)(HB + b) * CDIM * CDIM);
    __bf16* dst = (tid < 128) ? sA : sB;
    int r = tid & 127;
    const float* row = src + r * CDIM;
    float m = -3.4e38f;
    for (int j = 0; j < CDIM; ++j) m = fmaxf(m, row[j]);
    float sum = 0.f;
    for (int j = 0; j < CDIM; ++j) sum += __expf(row[j] - m);
    float inv = 1.f / sum;
    for (int j = 0; j < CDIM; ++j) dst[r * LDST + j] = (__bf16)(__expf(row[j] - m) * inv);
  }
  __syncthreads();

  // T = A_st @ A_ts^T  (contract over columns of both)
  const int wave = tid >> 5, lane = tid & 31, half = lane >> 4, lid = lane & 15;
  v8f acc[8];
#pragma unroll
  for (int dt = 0; dt < 8; ++dt) acc[dt] = (v8f){};
#pragma unroll
  for (int ks = 0; ks < 4; ++ks) {
    v16bf a = frag_ld(sA + (wave * 16 + lid) * LDST + ks * 32 + 8 * half);
#pragma unroll
    for (int dt = 0; dt < 8; ++dt) {
      v16bf bf = frag_ld(sB + (dt * 16 + lid) * LDST + ks * 32 + 8 * half);
      acc[dt] = wmma_bf16(a, bf, acc[dt]);
    }
  }
#pragma unroll
  for (int dt = 0; dt < 8; ++dt)
#pragma unroll
    for (int i = 0; i < 8; ++i)
      sT[(wave * 16 + half * 8 + i) * TS + dt * 16 + lid] = acc[dt][i];
  __syncthreads();

  if (tid < 128) {                              // final softmax -> att (bf16)
    float* row = sT + tid * TS;
    float m = -3.4e38f;
    for (int j = 0; j < CDIM; ++j) m = fmaxf(m, row[j]);
    float sum = 0.f;
    for (int j = 0; j < CDIM; ++j) sum += __expf(row[j] - m);
    float inv = 1.f / sum;
    gbf16* out = (gbf16*)(att + (size_t)b * CDIM * CDIM + tid * CDIM);
    for (int j = 0; j < CDIM; ++j) out[j] = (__bf16)(__expf(row[j] - m) * inv);
  }
}

// ---------------------------------------------------------------- output GEMM
// grid: (NPIX/128, NB).  out[g][o][n] = x[g][o][n] + sum_d att[g%8][o][d] V[g][d][n]
__global__ void __launch_bounds__(256)
k_out(const float* __restrict__ x, const __bf16* __restrict__ V,
      const __bf16* __restrict__ att, float* __restrict__ out) {
  extern __shared__ __align__(16) char smem[];
  __bf16* sAtt = (__bf16*)smem;              // [128][LDST]
  __bf16* sVT  = sAtt + CDIM * LDST;         // [128 n][LDST]  V tile transposed

  const int tid = threadIdx.x;
  const int g = blockIdx.y, bb = g & 7;
  const int n0 = blockIdx.x * 128;
  const __bf16* attb = att + (size_t)bb * CDIM * CDIM;
  const __bf16* Vb = V + (size_t)g * CDIM * NPIX;
  const float*  xb = x + (size_t)g * CDIM * NPIX;
  float*        ob = out + (size_t)g * CDIM * NPIX;

  const unsigned sAtt0 = lds_addr(sAtt);
  for (int i = tid; i < CDIM * 16; i += 256) {   // att tile: verbatim -> async
    int c = i >> 4, j4 = i & 15;
    async_cp_b128(sAtt0 + (unsigned)(c * LDST * 2 + j4 * 16),
                  attb + c * CDIM + j4 * 8);
  }
  for (int i = tid; i < CDIM * 128; i += 256) {  // V tile: transpose in VALU
    int d = i >> 7, j = i & 127;
    sVT[j * LDST + d] = Vb[(size_t)d * NPIX + n0 + j];
  }
  async_wait0();
  __syncthreads();

  const int wave = tid >> 5, lane = tid & 31, half = lane >> 4, lid = lane & 15;
  const __bf16* ab = sAtt + (wave * 16 + lid) * LDST + 8 * half;
  v16bf a0 = frag_ld(ab), a1 = frag_ld(ab + 32), a2 = frag_ld(ab + 64),
        a3 = frag_ld(ab + 96);
  for (int nt = 0; nt < 8; ++nt) {
    const __bf16* bb2 = sVT + (nt * 16 + lid) * LDST + 8 * half;
    v8f acc = {};
    acc = wmma_bf16(a0, frag_ld(bb2), acc);
    acc = wmma_bf16(a1, frag_ld(bb2 + 32), acc);
    acc = wmma_bf16(a2, frag_ld(bb2 + 64), acc);
    acc = wmma_bf16(a3, frag_ld(bb2 + 96), acc);
    int col = n0 + nt * 16 + lid;
#pragma unroll
    for (int i = 0; i < 8; ++i) {
      int o = wave * 16 + half * 8 + i;
      size_t idx = (size_t)o * NPIX + col;
      ob[idx] = xb[idx] + acc[i];                // residual fused
    }
  }
}

// ---------------------------------------------------------------- launcher
extern "C" void kernel_launch(void* const* d_in, const int* in_sizes, int n_in,
                              void* d_out, int out_size, void* d_ws, size_t ws_size,
                              hipStream_t stream) {
  (void)in_sizes; (void)n_in; (void)out_size; (void)ws_size;
  const float* x  = (const float*)d_in[0];
  const float* qw = (const float*)d_in[1];
  const float* qb = (const float*)d_in[2];
  const float* kw = (const float*)d_in[3];
  const float* kb = (const float*)d_in[4];
  const float* vw = (const float*)d_in[5];
  const float* vb = (const float*)d_in[6];

  const size_t bigBF = (size_t)NB * CDIM * NPIX;     // elems per Q/K/V buffer
  __bf16* Q = (__bf16*)d_ws;
  __bf16* K = Q + bigBF;
  __bf16* V = K + bigBF;
  float*  S = (float*)(V + bigBF);                   // [2][8][128][128] f32
  __bf16* att = (__bf16*)(S + 2 * HB * CDIM * CDIM); // [8][128][128] bf16
  float* out = (float*)d_out;

  const int sElems = 2 * HB * CDIM * CDIM;
  k_zero<<<(sElems + 255) / 256, 256, 0, stream>>>(S, sElems);

  dim3 gtile(NPIX / 128, NB);
  size_t lds_proj = (size_t)(3 * CDIM + CDIM) * LDST * sizeof(__bf16) +
                    3 * CDIM * sizeof(float);
  k_proj<<<gtile, 256, lds_proj, stream>>>(x, qw, qb, kw, kb, vw, vb, Q, K, V);

  dim3 ggram(16, HB, 2);
  size_t lds_gram = (size_t)2 * CDIM * LDST * sizeof(__bf16);
  k_gram<<<ggram, 256, lds_gram, stream>>>(Q, K, S);

  size_t lds_attn = (size_t)2 * CDIM * LDST * sizeof(__bf16) +
                    (size_t)CDIM * 129 * sizeof(float);
  k_attn<<<HB, 256, lds_attn, stream>>>(S, att);

  size_t lds_out = (size_t)2 * CDIM * LDST * sizeof(__bf16);
  k_out<<<gtile, 256, lds_out, stream>>>(x, V, att, out);
}
